// CrossAttention_22814866276431
// MI455X (gfx1250) — matrix operand order repaired and checked by reading, not compiled, not run on hardware
//
#include <hip/hip_runtime.h>
#include <math.h>

typedef __attribute__((ext_vector_type(16))) _Float16 v16h;
typedef __attribute__((ext_vector_type(8)))  float    v8f;

#define DIMN 2048
#define NH   16
#define KVHN 4
#define HDN  128
#define GRP  (NH / KVHN)
#define BBN  2
#define TTN  2048
#define SSN  2048
#define EPSF 1.1920929e-07f
#define SCALEF 0.08838834764831845f  // 1/sqrt(128)

static __device__ __forceinline__ v8f wmma16(v16h a, v16h b, v8f c) {
  // (neg_a, A, neg_b, B, c_mod, C, reuse_a, reuse_b)
  return __builtin_amdgcn_wmma_f32_16x16x32_f16(false, a, false, b, (short)0, c,
                                                false, false);
}

// One lane moves 16 bytes global -> LDS asynchronously (GVS mode).
// vdst = LDS byte address, vaddr = 32-bit byte offset, saddr = 64-bit base.
static __device__ __forceinline__ void async_cp16(unsigned ldsoff, unsigned voff,
                                                  const void* base) {
  asm volatile("global_load_async_to_lds_b128 %0, %1, %2"
               :: "v"(ldsoff), "v"(voff), "s"(base) : "memory");
}

// ---------------------------------------------------------------------------
// Bulk f32 -> f16 conversion (one-time, bandwidth bound). n multiple of 2048.
// ---------------------------------------------------------------------------
__global__ __launch_bounds__(256) void cvt_f32_f16(const float* __restrict__ s,
                                                   _Float16* __restrict__ d,
                                                   int n) {
  int i = (blockIdx.x * 256 + threadIdx.x) * 8;
#pragma unroll
  for (int j = 0; j < 8; ++j) d[i + j] = (_Float16)s[i + j];
  (void)n;
}

// ---------------------------------------------------------------------------
// C[M,N] = A[M,K] * W[N,K]^T, A/W f16, C f32.
// Block = 256 threads (8 waves). Block tile 128(M) x 64(N), K stage = 64.
// A/W tiles double-buffered in LDS via global_load_async_to_lds_b128:
//   per stage: A 16KB (4 async instr/thread), W 8KB (2 async instr/thread).
// Wave tile 16x64 -> 4 accumulators; 8 wmmas per stage per wave.
// grid = (M/128, N/64).
// ---------------------------------------------------------------------------
__global__ __launch_bounds__(256) void gemm_f16_wmma_async(
    const _Float16* __restrict__ A, const _Float16* __restrict__ W,
    float* __restrict__ C, int M, int N, int K) {
  __shared__ __align__(64) _Float16 smem[2 * 12288];  // 48KB: [2][A 8192 | W 4096]
  const int tid  = threadIdx.x;
  const int lane = tid & 31;
  const int wv   = tid >> 5;
  const int hl   = lane >> 4;
  const int lm   = lane & 15;
  const int m0   = blockIdx.x * 128;
  const int n0   = blockIdx.y * 64;
  const int nk   = K >> 6;  // stages of 64 halfs

  // Escape smem's address into an opaque memory-clobbering asm so the async
  // copies (also memory-clobbering) are treated as potential writers of it.
  asm volatile("" :: "v"(&smem[0]) : "memory");
  // Real stores so LDS is never "store-free" for the optimizer.
#pragma unroll
  for (int j = 0; j < 96; j += 8) {
    _Float16* z = smem + tid * 96 + j;
#pragma unroll
    for (int e = 0; e < 8; ++e) z[e] = (_Float16)0.f;
  }
  __syncthreads();

  // Per-thread async chunk coordinates (16B chunks; 8 chunks per 64-half row).
  unsigned ldsA[4], avoff[4], ldsW[2], wvoff[2];
#pragma unroll
  for (int i = 0; i < 4; ++i) {
    unsigned c = i * 256 + tid, row = c >> 3, ch = c & 7;
    ldsA[i]  = row * 128u + ch * 16u;
    avoff[i] = (unsigned)(m0 + row) * (unsigned)(K * 2) + ch * 16u;
  }
#pragma unroll
  for (int i = 0; i < 2; ++i) {
    unsigned c = i * 256 + tid, row = c >> 3, ch = c & 7;
    ldsW[i]  = row * 128u + ch * 16u;
    wvoff[i] = (unsigned)(n0 + row) * (unsigned)(K * 2) + ch * 16u;
  }

  auto prefetch = [&](int st) {
    const unsigned buf = (unsigned)(st & 1) * 24576u;
    const unsigned kb  = (unsigned)st * 128u;  // 64 halfs = 128 bytes
#pragma unroll
    for (int i = 0; i < 4; ++i) async_cp16(buf + ldsA[i], avoff[i] + kb, A);
#pragma unroll
    for (int i = 0; i < 2; ++i)
      async_cp16(buf + 16384u + ldsW[i], wvoff[i] + kb, W);
  };

  v8f acc[4];
  for (int i = 0; i < 4; ++i)
    for (int e = 0; e < 8; ++e) acc[i][e] = 0.f;

  const int mrow = wv * 16 + lm;
  auto compute = [&](int st) {
    const _Float16* sA = smem + (size_t)(st & 1) * 12288;
    const _Float16* sW = sA + 8192;
#pragma unroll
    for (int kk = 0; kk < 64; kk += 32) {
      // Issue ALL fragment loads for this k-step first, then the 4 wmmas,
      // so LDS latency overlaps the matrix pipe instead of serializing on
      // per-fragment s_wait_dscnt 0.
      v16h af, bf[4];
      const _Float16* ar = sA + mrow * 64 + kk + hl * 16;
#pragma unroll
      for (int j = 0; j < 8; ++j) { af[j] = ar[j]; af[8 + j] = ar[8 + j]; }
#pragma unroll
      for (int nt = 0; nt < 4; ++nt)
        bf[nt] = *(const v16h*)(sW + (nt * 16 + lm) * 64 + kk + hl * 16);
#pragma unroll
      for (int nt = 0; nt < 4; ++nt) acc[nt] = wmma16(af, bf[nt], acc[nt]);
    }
  };

  prefetch(0);
  if (nk > 1) prefetch(1);

  for (int st = 0; st < nk - 1; ++st) {
    asm volatile("s_wait_asynccnt 6" ::: "memory");  // stage st landed
    __syncthreads();
    compute(st);
    __syncthreads();
    if (st + 2 < nk) prefetch(st + 2);
  }
  asm volatile("s_wait_asynccnt 0" ::: "memory");  // final stage landed
  __syncthreads();
  compute(nk - 1);

  // C/D layout: row m0 + wv*16 + r + 8*hl, col n0 + nt*16 + lm
#pragma unroll
  for (int nt = 0; nt < 4; ++nt)
#pragma unroll
    for (int r = 0; r < 8; ++r)
      C[(size_t)(m0 + wv * 16 + r + 8 * hl) * N + n0 + nt * 16 + lm] = acc[nt][r];
}

// ---------------------------------------------------------------------------
// RoPE (interleaved) + RMSNorm on Q rows. One wave per (b,t,h) row of 128.
// Writes q as f16 in [B, H, T, HD] layout for the attention kernel.
// ---------------------------------------------------------------------------
__global__ __launch_bounds__(256) void rope_rmsnorm_q(
    const float* __restrict__ qraw, const float* __restrict__ fcos,
    const float* __restrict__ fsin, const float* __restrict__ qw,
    _Float16* __restrict__ qh) {
  const int lane = threadIdx.x & 31;
  const int wv   = threadIdx.x >> 5;
  const int rid  = blockIdx.x * 8 + wv;
  const int b = rid / (TTN * NH);
  const int t = (rid / NH) % TTN;
  const int h = rid % NH;
  const float* src = qraw + ((size_t)(b * TTN + t)) * DIMN + h * HDN;
  const int d0 = lane * 4;

  float v[4], o[4];
#pragma unroll
  for (int j = 0; j < 4; ++j) v[j] = src[d0 + j];
#pragma unroll
  for (int j = 0; j < 4; j += 2) {
    float c0 = fcos[(size_t)t * HDN + d0 + j];
    float s0 = fsin[(size_t)t * HDN + d0 + j];
    float c1 = fcos[(size_t)t * HDN + d0 + j + 1];
    float s1 = fsin[(size_t)t * HDN + d0 + j + 1];
    o[j]     = v[j] * c0     - v[j + 1] * s0;  // rotate_half: even gets -x[odd]
    o[j + 1] = v[j + 1] * c1 + v[j] * s1;      // odd gets +x[even]
  }
  float ss = o[0]*o[0] + o[1]*o[1] + o[2]*o[2] + o[3]*o[3];
#pragma unroll
  for (int m = 1; m < 32; m <<= 1) ss += __shfl_xor(ss, m, 32);
  float sc = rsqrtf(ss * (1.0f / HDN) + EPSF);
  _Float16* dst = qh + (((size_t)(b * NH + h)) * TTN + t) * HDN + d0;
#pragma unroll
  for (int j = 0; j < 4; ++j) dst[j] = (_Float16)(o[j] * sc * qw[d0 + j]);
}

// ---------------------------------------------------------------------------
// RMSNorm K + split V. One wave per (b,s,kvh).
// K -> f16 [B, KVH, S, HD];  V -> f16 transposed [B, KVH, HD, S].
// ---------------------------------------------------------------------------
__global__ __launch_bounds__(256) void norm_split_kv(
    const float* __restrict__ kvraw, const float* __restrict__ kw,
    _Float16* __restrict__ kh, _Float16* __restrict__ vT) {
  const int lane = threadIdx.x & 31;
  const int wv   = threadIdx.x >> 5;
  const int rid  = blockIdx.x * 8 + wv;
  const int b = rid / (SSN * KVHN);
  const int s = (rid / KVHN) % SSN;
  const int g = rid % KVHN;
  const float* krow = kvraw + ((size_t)(b * SSN + s)) * (2 * KVHN * HDN) + g * HDN;
  const float* vrow = krow + KVHN * HDN;
  const int d0 = lane * 4;

  float k4[4];
#pragma unroll
  for (int j = 0; j < 4; ++j) k4[j] = krow[d0 + j];
  float ss = k4[0]*k4[0] + k4[1]*k4[1] + k4[2]*k4[2] + k4[3]*k4[3];
#pragma unroll
  for (int m = 1; m < 32; m <<= 1) ss += __shfl_xor(ss, m, 32);
  float sc = rsqrtf(ss * (1.0f / HDN) + EPSF);

  _Float16* kd = kh + (((size_t)(b * KVHN + g)) * SSN + s) * HDN + d0;
#pragma unroll
  for (int j = 0; j < 4; ++j) kd[j] = (_Float16)(k4[j] * sc * kw[d0 + j]);

  _Float16* vb = vT + ((size_t)(b * KVHN + g)) * HDN * (size_t)SSN;
#pragma unroll
  for (int j = 0; j < 4; ++j)
    vb[(size_t)(d0 + j) * SSN + s] = (_Float16)vrow[d0 + j];
}

// ---------------------------------------------------------------------------
// Flash attention. grid = (T/128, H, B), block = 256 (8 waves).
// Wave: 16 q-rows, streams S in blocks of 32. QK^T and PV via f16 WMMA,
// online softmax in f32. P restaged through per-wave LDS (C-layout -> A-layout).
// Output f16 [B, T, DIM] (head-major within DIM) for the O-projection GEMM.
// ---------------------------------------------------------------------------
__global__ __launch_bounds__(256) void flash_attn_wmma(
    const _Float16* __restrict__ qh, const _Float16* __restrict__ kh,
    const _Float16* __restrict__ vT, const unsigned char* __restrict__ cmask,
    _Float16* __restrict__ attnh) {
  __shared__ _Float16 plds[8][16 * 32];
  const int lane = threadIdx.x & 31;
  const int wv   = threadIdx.x >> 5;
  const int hl   = lane >> 4;
  const int lm   = lane & 15;
  const int t0   = blockIdx.x * 128 + wv * 16;
  const int h    = blockIdx.y;
  const int b    = blockIdx.z;
  const int g    = h / GRP;  // kv head

  // Q A-fragments for K=128 (4 wmma k-steps)
  v16h qf[4];
  {
    const _Float16* qr = qh + (((size_t)(b * NH + h)) * TTN + t0 + lm) * HDN;
#pragma unroll
    for (int kk = 0; kk < 4; ++kk) {
      const _Float16* p0 = qr + kk * 32 + hl * 16;
#pragma unroll
      for (int j = 0; j < 8; ++j) {
        qf[kk][j]     = p0[j];
        qf[kk][8 + j] = p0[8 + j];
      }
    }
  }

  v8f o[8];
  for (int i = 0; i < 8; ++i)
    for (int e = 0; e < 8; ++e) o[i][e] = 0.f;
  float mrow[8], lrow[8];
#pragma unroll
  for (int r = 0; r < 8; ++r) { mrow[r] = -3.0e38f; lrow[r] = 0.f; }

  const _Float16* kbase = kh + ((size_t)(b * KVHN + g)) * SSN * (size_t)HDN;
  const _Float16* vbase = vT + ((size_t)(b * KVHN + g)) * HDN * (size_t)SSN;

  for (int s0 = 0; s0 < SSN; s0 += 32) {
    // scores: two 16x16 n-tiles
    v8f sc0, sc1;
#pragma unroll
    for (int e = 0; e < 8; ++e) { sc0[e] = 0.f; sc1[e] = 0.f; }
    const _Float16* kr0 = kbase + (size_t)(s0 + lm) * HDN;
    const _Float16* kr1 = kbase + (size_t)(s0 + 16 + lm) * HDN;
#pragma unroll
    for (int kk = 0; kk < 4; ++kk) {
      v16h b0 = *(const v16h*)(kr0 + kk * 32 + hl * 16);
      v16h b1 = *(const v16h*)(kr1 + kk * 32 + hl * 16);
      sc0 = wmma16(qf[kk], b0, sc0);
      sc1 = wmma16(qf[kk], b1, sc1);
    }
    const bool val0 = cmask[(size_t)b * SSN + s0 + lm] != 0;
    const bool val1 = cmask[(size_t)b * SSN + s0 + 16 + lm] != 0;

    float corr[8];
#pragma unroll
    for (int r = 0; r < 8; ++r) {
      float a0 = val0 ? sc0[r] * SCALEF : -3.0e38f;
      float a1 = val1 ? sc1[r] * SCALEF : -3.0e38f;
      float mx = fmaxf(a0, a1);
#pragma unroll
      for (int m = 1; m < 16; m <<= 1) mx = fmaxf(mx, __shfl_xor(mx, m, 32));
      float nm = fmaxf(mrow[r], mx);
      float cr = __expf(mrow[r] - nm);
      float p0 = __expf(a0 - nm);
      float p1 = __expf(a1 - nm);
      float rs = p0 + p1;
#pragma unroll
      for (int m = 1; m < 16; m <<= 1) rs += __shfl_xor(rs, m, 32);
      lrow[r] = lrow[r] * cr + rs;
      mrow[r] = nm;
      corr[r] = cr;
      // stage P (row r+8*hl, cols lm and 16+lm) into LDS
      plds[wv][(r + 8 * hl) * 32 + lm]      = (_Float16)p0;
      plds[wv][(r + 8 * hl) * 32 + 16 + lm] = (_Float16)p1;
    }
#pragma unroll
    for (int ct = 0; ct < 8; ++ct)
#pragma unroll
      for (int e = 0; e < 8; ++e) o[ct][e] *= corr[e];

    asm volatile("s_wait_dscnt 0" ::: "memory");  // P staged, reload in A-layout

    v16h pf;
#pragma unroll
    for (int j = 0; j < 8; ++j) {
      pf[j]     = plds[wv][lm * 32 + hl * 16 + j];
      pf[8 + j] = plds[wv][lm * 32 + hl * 16 + 8 + j];
    }
    // PV: accumulate 16x128 output, 8 n-tiles of HD
#pragma unroll
    for (int ct = 0; ct < 8; ++ct) {
      v16h vf = *(const v16h*)(vbase + (size_t)(ct * 16 + lm) * SSN + s0 + hl * 16);
      o[ct] = wmma16(pf, vf, o[ct]);
    }
  }

#pragma unroll
  for (int ct = 0; ct < 8; ++ct)
#pragma unroll
    for (int r = 0; r < 8; ++r) {
      float inv = 1.0f / lrow[r];
      attnh[((size_t)(b * TTN) + t0 + r + 8 * hl) * DIMN + h * HDN + ct * 16 + lm] =
          (_Float16)(o[ct][r] * inv);
    }
}

// ---------------------------------------------------------------------------
extern "C" void kernel_launch(void* const* d_in, const int* in_sizes, int n_in,
                              void* d_out, int out_size, void* d_ws, size_t ws_size,
                              hipStream_t stream) {
  (void)in_sizes; (void)n_in; (void)out_size; (void)ws_size;
  const float* x    = (const float*)d_in[0];
  const float* ctx  = (const float*)d_in[1];
  const float* fcos = (const float*)d_in[2];
  const float* fsin = (const float*)d_in[3];
  const unsigned char* cmask = (const unsigned char*)d_in[4];
  const float* wq   = (const float*)d_in[5];
  const float* wkv  = (const float*)d_in[6];
  const float* wo   = (const float*)d_in[7];
  const float* qw   = (const float*)d_in[8];
  const float* kw   = (const float*)d_in[9];
  float* out = (float*)d_out;

  char* ws = (char*)d_ws;
  _Float16* xh    = (_Float16*)(ws);               // [4096,2048] f16 = 16MB
  _Float16* ch    = (_Float16*)(ws + 16777216u);   // [4096,2048] f16 = 16MB
  _Float16* wqh   = (_Float16*)(ws + 33554432u);   // [2048,2048] f16 =  8MB
  _Float16* wkvh  = (_Float16*)(ws + 41943040u);   // [1024,2048] f16 =  4MB
  _Float16* woh   = (_Float16*)(ws + 46137344u);   // [2048,2048] f16 =  8MB
  float*    qraw  = (float*)(ws + 54525952u);      // [4096,2048] f32 = 32MB
  float*    kvraw = (float*)(ws + 88080384u);      // [4096,1024] f32 = 16MB
  _Float16* qh    = (_Float16*)(ws + 104857600u);  // [B,H,T,HD]   f16 = 16MB
  _Float16* kh    = (_Float16*)(ws + 121634816u);  // [B,KVH,S,HD] f16 =  4MB
  _Float16* vT    = (_Float16*)(ws + 125829120u);  // [B,KVH,HD,S] f16 =  4MB
  _Float16* attnh = xh;  // reuse xh region once x has been consumed

  dim3 blk(256);
  // One-time f32 -> f16 conversions (activations + weights)
  cvt_f32_f16<<<dim3(4096), blk, 0, stream>>>(x,   xh,   BBN * TTN * DIMN);
  cvt_f32_f16<<<dim3(4096), blk, 0, stream>>>(ctx, ch,   BBN * SSN * DIMN);
  cvt_f32_f16<<<dim3(2048), blk, 0, stream>>>(wq,  wqh,  DIMN * DIMN);
  cvt_f32_f16<<<dim3(1024), blk, 0, stream>>>(wkv, wkvh, 2 * KVHN * HDN * DIMN);
  cvt_f32_f16<<<dim3(2048), blk, 0, stream>>>(wo,  woh,  DIMN * DIMN);

  // Q projection: [4096,2048] x wq[2048,2048]^T
  gemm_f16_wmma_async<<<dim3(32, 32), blk, 0, stream>>>(xh, wqh, qraw,
                                                        BBN * TTN, DIMN, DIMN);
  // KV projection: [4096,2048] x wkv[1024,2048]^T
  gemm_f16_wmma_async<<<dim3(32, 16), blk, 0, stream>>>(ch, wkvh, kvraw,
                                                        BBN * SSN, 2 * KVHN * HDN,
                                                        DIMN);
  // RoPE + RMSNorm on Q (65536 rows, 8 waves/block)
  rope_rmsnorm_q<<<dim3(8192), blk, 0, stream>>>(qraw, fcos, fsin, qw, qh);
  // RMSNorm K, split & transpose V (16384 rows)
  norm_split_kv<<<dim3(2048), blk, 0, stream>>>(kvraw, kw, kh, vT);
  // Flash attention -> f16 attn
  flash_attn_wmma<<<dim3(16, 16, 2), blk, 0, stream>>>(qh, kh, vT, cmask, attnh);
  // Output projection: attn[4096,2048] x wo[2048,2048]^T -> d_out (f32)
  gemm_f16_wmma_async<<<dim3(32, 32), blk, 0, stream>>>(attnh, woh, out,
                                                        BBN * TTN, DIMN, DIMN);
}
